// FilterMU_Unite_39247411151422
// MI455X (gfx1250) — compile-verified, matrix operand order checked by reading, not run on hardware
//
#include <hip/hip_runtime.h>

// CDNA5 / gfx1250: wave32, WMMA 16x16x32 bf16 -> f32 accumulate.
typedef __attribute__((ext_vector_type(16))) __bf16 v16bf;
typedef __attribute__((ext_vector_type(8)))  float  v8f;

__device__ __forceinline__ float softplus_f(float x) {
  // log(1 + exp(x)), numerically stable (matches jax.nn.softplus)
  return fmaxf(x, 0.0f) + log1pf(expf(-fabsf(x)));
}

// -----------------------------------------------------------------------------
// Generic GEMM: C[M,N] = act(A[M,K] @ B[K,N] + bias), K compile-time.
// A given as two stacked row-major blocks: rows [0,splitM) from A0,
// rows [splitM,M) from A1 (A1 may be null when splitM >= M). B row-major.
// One wave computes one 16x32 strip of C (two 16x16 N-tiles sharing one A
// fragment) with v_wmma_f32_16x16x32_bf16; f32 operands are converted to
// bf16 in registers while gathering. K fully unrolled -> 2*K/32 static WMMAs
// issued back-to-back (XDL pipe throughput mode).
// Preconditions (all call sites satisfy them): M%16==0, N%32==0, K%32==0,
// and (M/16)*(N/32) % (blockDim/32) == 0 so EXEC is all-ones for every WMMA.
// -----------------------------------------------------------------------------
template <int K>
__global__ void gemm_bf16_wmma_kernel(const float* __restrict__ A0,
                                      const float* __restrict__ A1,
                                      int splitM, int lda,
                                      const float* __restrict__ B, int ldb,
                                      const float* __restrict__ bias,
                                      float* __restrict__ C, int ldc,
                                      int N, int do_relu) {
  const int wavesPerBlock = blockDim.x >> 5;
  const int wave = blockIdx.x * wavesPerBlock + (threadIdx.x >> 5);
  const int lane = threadIdx.x & 31;
  const int stripsN = N >> 5;             // strips of 32 columns
  const int m0 = (wave / stripsN) << 4;
  const int n0 = (wave % stripsN) << 5;
  const int g  = lane >> 4;   // half-wave selector (ISA K-group +8 / M-row +8)
  const int mr = lane & 15;   // A row / B,C column within the tile

  const int arow = m0 + mr;
  const float* __restrict__ Arow =
      (arow < splitM) ? (A0 + (size_t)arow * lda)
                      : (A1 + (size_t)(arow - splitM) * lda);

  v8f acc0 = {};
  v8f acc1 = {};
#pragma unroll
  for (int kk = 0; kk < K; kk += 32) {
    if (kk + 32 < K) __builtin_prefetch(Arow + kk + 32, 0, 0);  // global_prefetch_b8
    v16bf afrag, bfrag0, bfrag1;
#pragma unroll
    for (int v = 0; v < 8; ++v) {
      // 16-bit operand layout (ISA 7.12.2): VGPR v holds K pair
      //   k0 = (v&3)*2 + (v>>2)*16 + (lane>=16 ? 8 : 0)
      const int k0 = kk + ((v & 3) << 1) + ((v >> 2) << 4) + (g << 3);
      afrag[2 * v + 0] = (__bf16)Arow[k0 + 0];
      afrag[2 * v + 1] = (__bf16)Arow[k0 + 1];
      const float* __restrict__ b0 = B + (size_t)k0 * ldb + n0 + mr;
      const float* __restrict__ b1 = b0 + ldb;
      bfrag0[2 * v + 0] = (__bf16)b0[0];
      bfrag0[2 * v + 1] = (__bf16)b1[0];
      bfrag1[2 * v + 0] = (__bf16)b0[16];
      bfrag1[2 * v + 1] = (__bf16)b1[16];
    }
    acc0 = __builtin_amdgcn_wmma_f32_16x16x32_bf16(
        false, afrag, false, bfrag0, (short)0, acc0, false, false);
    acc1 = __builtin_amdgcn_wmma_f32_16x16x32_bf16(
        false, afrag, false, bfrag1, (short)0, acc1, false, false);
  }

  const float bv0 = bias ? bias[n0 + mr] : 0.0f;
  const float bv1 = bias ? bias[n0 + 16 + mr] : 0.0f;
#pragma unroll
  for (int r = 0; r < 8; ++r) {
    // C layout (ISA 7.12.2): VGPR r -> row m0 + r + 8*g, col n0 + (lane&15)
    const int row = m0 + r + (g << 3);
    float v0 = acc0[r] + bv0;
    float v1 = acc1[r] + bv1;
    if (do_relu) { v0 = fmaxf(v0, 0.0f); v1 = fmaxf(v1, 0.0f); }
    C[(size_t)row * ldc + n0 + mr]      = v0;
    C[(size_t)row * ldc + n0 + 16 + mr] = v1;
  }
}

// -----------------------------------------------------------------------------
// Pairwise critic head + row-wise logsumexp.
//   T1[i,j] = softplus( sum_k relu(Hyb[i,k] + Hx[j,k]) * W2[k] + b2 )
//   lse[i]  = logsumexp_j T1[i,j];   t0[i] = T1[i,i]  (== F(x_i, y_i))
// One block per row i; NCOL threads, thread j owns column j.
// Hyb already contains y@W1y + b1. Never materializes [N,N,128].
// -----------------------------------------------------------------------------
template <int NCOL>
__global__ void pairhead_lse_kernel(const float* __restrict__ Hx,   // [NCOL,128]
                                    const float* __restrict__ Hyb,  // [nrow,128]
                                    const float* __restrict__ W2,   // [128]
                                    const float* __restrict__ b2,   // [1]
                                    float* __restrict__ lse,        // [nrow]
                                    float* __restrict__ t0) {       // [nrow]
  __shared__ float sHy[128];
  __shared__ float sW2[128];
  __shared__ float sred[NCOL];
  const int i = blockIdx.x;
  const int j = threadIdx.x;
  if (j < 128) { sHy[j] = Hyb[i * 128 + j]; sW2[j] = W2[j]; }
  __syncthreads();

  const float* __restrict__ hx = Hx + (size_t)j * 128;
  float s = 0.0f;
#pragma unroll 8
  for (int k = 0; k < 128; ++k)
    s = fmaf(fmaxf(sHy[k] + hx[k], 0.0f), sW2[k], s);
  const float t1 = softplus_f(s + b2[0]);
  if (j == i) t0[i] = t1;

  // row max (deterministic tree)
  sred[j] = t1;
  __syncthreads();
  for (int str = NCOL >> 1; str > 0; str >>= 1) {
    if (j < str) sred[j] = fmaxf(sred[j], sred[j + str]);
    __syncthreads();
  }
  const float mx = sred[0];
  __syncthreads();

  // sum exp(t1 - max)
  sred[j] = expf(t1 - mx);
  __syncthreads();
  for (int str = NCOL >> 1; str > 0; str >>= 1) {
    if (j < str) sred[j] += sred[j + str];
    __syncthreads();
  }
  if (j == 0) lse[i] = mx + logf(sred[0]);
}

// -----------------------------------------------------------------------------
// Final scalar combine:
//   loss_k = mean(t0) - (mean(lse) - log(n)) = mean(t0 - lse) + log(n)
//   out    = -loss2 + lam * loss1
// -----------------------------------------------------------------------------
__global__ void finalize_kernel(const float* __restrict__ lse2,
                                const float* __restrict__ t02,   // n=256
                                const float* __restrict__ lse1,
                                const float* __restrict__ t01,   // n=512
                                const float* __restrict__ lam,
                                float* __restrict__ out) {
  __shared__ float s1[512];
  __shared__ float s2[512];
  const int t = threadIdx.x;  // 512 threads
  s1[t] = t01[t] - lse1[t];
  s2[t] = (t < 256) ? (t02[t] - lse2[t]) : 0.0f;
  __syncthreads();
  for (int str = 256; str > 0; str >>= 1) {
    if (t < str) { s1[t] += s1[t + str]; s2[t] += s2[t + str]; }
    __syncthreads();
  }
  if (t == 0) {
    const float loss1 = s1[0] / 512.0f + logf(512.0f);
    const float loss2 = s2[0] / 256.0f + logf(256.0f);
    out[0] = -loss2 + lam[0] * loss1;
  }
}

extern "C" void kernel_launch(void* const* d_in, const int* in_sizes, int n_in,
                              void* d_out, int out_size, void* d_ws, size_t ws_size,
                              hipStream_t stream) {
  (void)in_sizes; (void)n_in; (void)out_size; (void)ws_size;
  const float* sent_a = (const float*)d_in[0];   // [256,256]
  const float* word_a = (const float*)d_in[1];   // [256,256]
  const float* sent_b = (const float*)d_in[2];   // [256,256]
  const float* word_b = (const float*)d_in[3];   // [256,256]
  const float* lam    = (const float*)d_in[4];   // [1]
  const float* We     = (const float*)d_in[5];   // [256,256]
  const float* be     = (const float*)d_in[6];   // [256]
  const float* Wf1    = (const float*)d_in[7];   // [512,128]
  const float* bf1    = (const float*)d_in[8];   // [128]
  const float* Wf2    = (const float*)d_in[9];   // [128,1]
  const float* bf2    = (const float*)d_in[10];  // [1]
  const float* Wc1    = (const float*)d_in[11];  // [512,128]
  const float* bc1    = (const float*)d_in[12];  // [128]
  const float* Wc2    = (const float*)d_in[13];  // [128,1]
  const float* bc2    = (const float*)d_in[14];  // [1]

  // Workspace layout (floats); total ~1.3 MB — lives in L2.
  float* ws     = (float*)d_ws;
  float* debias = ws;  ws += 512 * 256;   // rows 0-255: debias_a, 256-511: debias_b
  float* Hx2    = ws;  ws += 256 * 128;
  float* Hy2    = ws;  ws += 256 * 128;
  float* Hx1    = ws;  ws += 512 * 128;
  float* Hy1    = ws;  ws += 512 * 128;
  float* lse2   = ws;  ws += 256;
  float* t02    = ws;  ws += 256;
  float* lse1   = ws;  ws += 512;
  float* t01    = ws;  ws += 512;

  const int BLK = 128;  // 4 waves per block; all strip counts divide by 4

  // 1) debias = relu([sent_a; sent_b] @ We + be)       [512,256], 256 strips
  gemm_bf16_wmma_kernel<256><<<256 / 4, BLK, 0, stream>>>(
      sent_a, sent_b, 256, 256, We, 256, be, debias, 256, 256, 1);

  // 2) Hx2 = debias_a @ Wf1[:256]                      [256,128], 64 strips
  gemm_bf16_wmma_kernel<256><<<64 / 4, BLK, 0, stream>>>(
      debias, nullptr, 512, 256, Wf1, 128, nullptr, Hx2, 128, 128, 0);

  // 3) Hy2 = debias_b @ Wf1[256:] + bf1                [256,128], 64 strips
  gemm_bf16_wmma_kernel<256><<<64 / 4, BLK, 0, stream>>>(
      debias + 256 * 256, nullptr, 512, 256, Wf1 + 256 * 128, 128, bf1,
      Hy2, 128, 128, 0);

  // 4) Hx1 = debias @ Wc1[:256]                        [512,128], 128 strips
  gemm_bf16_wmma_kernel<256><<<128 / 4, BLK, 0, stream>>>(
      debias, nullptr, 512, 256, Wc1, 128, nullptr, Hx1, 128, 128, 0);

  // 5) Hy1 = [word_a; word_b] @ Wc1[256:] + bc1        [512,128], 128 strips
  gemm_bf16_wmma_kernel<256><<<128 / 4, BLK, 0, stream>>>(
      word_a, word_b, 256, 256, Wc1 + 256 * 128, 128, bc1, Hy1, 128, 128, 0);

  // 6) pairwise heads + row-wise logsumexp (T0 = diagonal of T1)
  pairhead_lse_kernel<256><<<256, 256, 0, stream>>>(Hx2, Hy2, Wf2, bf2, lse2, t02);
  pairhead_lse_kernel<512><<<512, 512, 0, stream>>>(Hx1, Hy1, Wc2, bc2, lse1, t01);

  // 7) out = -loss2 + lam * loss1
  finalize_kernel<<<1, 512, 0, stream>>>(lse2, t02, lse1, t01, lam, (float*)d_out);
}
